// RealImagFiveValLinear_73555609911441
// MI455X (gfx1250) — matrix operand order repaired and checked by reading, not compiled, not run on hardware
//
#include <hip/hip_runtime.h>

typedef __attribute__((ext_vector_type(16))) _Float16       v16h;
typedef __attribute__((ext_vector_type(8)))  float          v8f;

// Problem constants (fixed-shape task)
constexpr int M_TOK = 8192;
constexpr int N_OUT = 4096;
constexpr int K_IN  = 4096;
constexpr int BM = 128, BN = 128, BK = 32;
constexpr int LDH = 40;                 // LDS row stride in halves (80B): 16B aligned, bank-conflict free
constexpr long W_ELEMS = (long)N_OUT * K_IN;   // 16,777,216
constexpr long W_F4    = W_ELEMS / 4;          // float4 count

// ---------------------------------------------------------------------------
// Kernel 1: per-block partial abs-sums of both weight matrices (deterministic)
// partials layout: [0..511] = re, [512..1023] = im
// ---------------------------------------------------------------------------
__global__ __launch_bounds__(256) void abssum_kernel(const float* __restrict__ wre,
                                                     const float* __restrict__ wim,
                                                     float* __restrict__ partials) {
  __shared__ float sre[256];
  __shared__ float sim[256];
  const int tid = threadIdx.x;
  float ar = 0.0f, ai = 0.0f;
  for (long i = (long)blockIdx.x * 256 + tid; i < W_F4; i += (long)gridDim.x * 256) {
    float4 r = ((const float4*)wre)[i];
    ar += fabsf(r.x) + fabsf(r.y) + fabsf(r.z) + fabsf(r.w);
    float4 m = ((const float4*)wim)[i];
    ai += fabsf(m.x) + fabsf(m.y) + fabsf(m.z) + fabsf(m.w);
  }
  sre[tid] = ar; sim[tid] = ai;
  __syncthreads();
  for (int s = 128; s > 0; s >>= 1) {
    if (tid < s) { sre[tid] += sre[tid + s]; sim[tid] += sim[tid + s]; }
    __syncthreads();
  }
  if (tid == 0) {
    partials[blockIdx.x]       = sre[0];
    partials[512 + blockIdx.x] = sim[0];
  }
}

// ---------------------------------------------------------------------------
// Kernel 2: finalize scales.  params = {2/scale_re, scale_re/2, 2/scale_im, scale_im/2}
// ---------------------------------------------------------------------------
__global__ __launch_bounds__(256) void finalize_kernel(const float* __restrict__ partials,
                                                       float* __restrict__ params) {
  __shared__ float sre[256];
  __shared__ float sim[256];
  const int t = threadIdx.x;
  sre[t] = partials[t] + partials[t + 256];
  sim[t] = partials[512 + t] + partials[512 + t + 256];
  __syncthreads();
  for (int s = 128; s > 0; s >>= 1) {
    if (t < s) { sre[t] += sre[t + s]; sim[t] += sim[t + s]; }
    __syncthreads();
  }
  if (t == 0) {
    const float nrm = 1.0f / (float)W_ELEMS;
    const float sc_re = fmaxf(sre[0] * nrm, 1e-5f);
    const float sc_im = fmaxf(sim[0] * nrm, 1e-5f);
    params[0] = 2.0f / sc_re;  params[1] = 0.5f * sc_re;
    params[2] = 2.0f / sc_im;  params[3] = 0.5f * sc_im;
  }
}

// ---------------------------------------------------------------------------
// Kernel 3: fused quantize + complex GEMM via v_wmma_f32_16x16x32_f16
//   y_real = x_re@Wre^T + x_im@(-Wim)^T ;  y_imag = x_re@Wim^T + x_im@Wre^T
// Block tile 128x128, 8 waves (4x2), wave tile 32x64 (2x4 16x16 tiles).
// Register-staged pipeline: global loads of tile k+1 fly under WMMAs of tile k.
// LDS holds B as {Wre, Wim, -Wim} so the WMMA phase needs no VALU (no hazards).
// ---------------------------------------------------------------------------
__global__ __launch_bounds__(256) void cgemm_q_kernel(const float* __restrict__ xre,
                                                      const float* __restrict__ xim,
                                                      const float* __restrict__ wre,
                                                      const float* __restrict__ wim,
                                                      const float* __restrict__ params,
                                                      float* __restrict__ out) {
  __shared__ alignas(16) _Float16 lA[2][BM][LDH];  // [xre/xim][row m][k]
  __shared__ alignas(16) _Float16 lB[3][BN][LDH];  // [wre/wim/-wim][row n][k]

  const int tid  = threadIdx.x;
  const int lane = tid & 31;
  const int wave = tid >> 5;
  const int wm   = wave >> 1;           // 0..3
  const int wn   = wave & 1;            // 0..1
  const int bm   = blockIdx.y * BM;
  const int bn   = blockIdx.x * BN;

  const float inv2_re = params[0], hsc_re = params[1];
  const float inv2_im = params[2], hsc_im = params[3];

  const int lrow = tid >> 1;            // 0..127: tile row this thread loads
  const int kloc = (tid & 1) * 16;      // 0 or 16: K half

  const v8f vz = {0.f,0.f,0.f,0.f,0.f,0.f,0.f,0.f};
  v8f accR[2][4], accI[2][4];
#pragma unroll
  for (int i = 0; i < 2; ++i)
#pragma unroll
    for (int j = 0; j < 4; ++j) { accR[i][j] = vz; accI[i][j] = vz; }

  const int l15 = lane & 15;
  const int lhi = lane >> 4;

  union FragH { v16h v; uint4 q[2]; };

  // ---- register staging for the software pipeline ----
  float4 sxr[4], sxi[4], swr[4], swi[4];

  const size_t rowA = (size_t)(bm + lrow) * K_IN + kloc;
  const size_t rowB = (size_t)(bn + lrow) * K_IN + kloc;

  // prologue: stage tile k0 = 0
#pragma unroll
  for (int q = 0; q < 4; ++q) {
    sxr[q] = *(const float4*)(xre + rowA + 4 * q);
    sxi[q] = *(const float4*)(xim + rowA + 4 * q);
    swr[q] = *(const float4*)(wre + rowB + 4 * q);
    swi[q] = *(const float4*)(wim + rowB + 4 * q);
  }

  for (int k0 = 0; k0 < K_IN; k0 += BK) {
    __syncthreads();  // previous iteration's fragment reads complete

    // ---- convert staged tile into LDS ----
    {
      alignas(16) _Float16 t[16];
#pragma unroll
      for (int q = 0; q < 4; ++q) {
        t[4*q+0] = (_Float16)sxr[q].x; t[4*q+1] = (_Float16)sxr[q].y;
        t[4*q+2] = (_Float16)sxr[q].z; t[4*q+3] = (_Float16)sxr[q].w;
      }
      *(uint4*)&lA[0][lrow][kloc]     = *(const uint4*)&t[0];
      *(uint4*)&lA[0][lrow][kloc + 8] = *(const uint4*)&t[8];
#pragma unroll
      for (int q = 0; q < 4; ++q) {
        t[4*q+0] = (_Float16)sxi[q].x; t[4*q+1] = (_Float16)sxi[q].y;
        t[4*q+2] = (_Float16)sxi[q].z; t[4*q+3] = (_Float16)sxi[q].w;
      }
      *(uint4*)&lA[1][lrow][kloc]     = *(const uint4*)&t[0];
      *(uint4*)&lA[1][lrow][kloc + 8] = *(const uint4*)&t[8];
      // W_re: quantize to {0,±0.5,±1}*scale
#pragma unroll
      for (int q = 0; q < 4; ++q) {
        float v0 = fminf(2.f, fmaxf(-2.f, __builtin_rintf(swr[q].x * inv2_re)));
        float v1 = fminf(2.f, fmaxf(-2.f, __builtin_rintf(swr[q].y * inv2_re)));
        float v2 = fminf(2.f, fmaxf(-2.f, __builtin_rintf(swr[q].z * inv2_re)));
        float v3 = fminf(2.f, fmaxf(-2.f, __builtin_rintf(swr[q].w * inv2_re)));
        t[4*q+0] = (_Float16)(v0 * hsc_re); t[4*q+1] = (_Float16)(v1 * hsc_re);
        t[4*q+2] = (_Float16)(v2 * hsc_re); t[4*q+3] = (_Float16)(v3 * hsc_re);
      }
      *(uint4*)&lB[0][lrow][kloc]     = *(const uint4*)&t[0];
      *(uint4*)&lB[0][lrow][kloc + 8] = *(const uint4*)&t[8];
      // W_im (+ negated copy, exact sign flip at f32 level)
      alignas(16) _Float16 tn[16];
#pragma unroll
      for (int q = 0; q < 4; ++q) {
        float v0 = fminf(2.f, fmaxf(-2.f, __builtin_rintf(swi[q].x * inv2_im)));
        float v1 = fminf(2.f, fmaxf(-2.f, __builtin_rintf(swi[q].y * inv2_im)));
        float v2 = fminf(2.f, fmaxf(-2.f, __builtin_rintf(swi[q].z * inv2_im)));
        float v3 = fminf(2.f, fmaxf(-2.f, __builtin_rintf(swi[q].w * inv2_im)));
        float w0 = v0 * hsc_im, w1 = v1 * hsc_im, w2 = v2 * hsc_im, w3 = v3 * hsc_im;
        t[4*q+0]  = (_Float16)w0;  t[4*q+1]  = (_Float16)w1;
        t[4*q+2]  = (_Float16)w2;  t[4*q+3]  = (_Float16)w3;
        tn[4*q+0] = (_Float16)-w0; tn[4*q+1] = (_Float16)-w1;
        tn[4*q+2] = (_Float16)-w2; tn[4*q+3] = (_Float16)-w3;
      }
      *(uint4*)&lB[1][lrow][kloc]     = *(const uint4*)&t[0];
      *(uint4*)&lB[1][lrow][kloc + 8] = *(const uint4*)&t[8];
      *(uint4*)&lB[2][lrow][kloc]     = *(const uint4*)&tn[0];
      *(uint4*)&lB[2][lrow][kloc + 8] = *(const uint4*)&tn[8];
    }
    __syncthreads();

    // ---- stage next tile: loads in flight during the WMMA phase below ----
    if (k0 + BK < K_IN) {
      const size_t ga = rowA + k0 + BK;
      const size_t gb = rowB + k0 + BK;
#pragma unroll
      for (int q = 0; q < 4; ++q) {
        sxr[q] = *(const float4*)(xre + ga + 4 * q);
        sxi[q] = *(const float4*)(xim + ga + 4 * q);
        swr[q] = *(const float4*)(wre + gb + 4 * q);
        swi[q] = *(const float4*)(wim + gb + 4 * q);
      }
      if (k0 + 2 * BK < K_IN) {  // L2 prefetch for the streamed X panels
        __builtin_prefetch(xre + ga + BK, 0, 1);
        __builtin_prefetch(xim + ga + BK, 0, 1);
      }
    }

    // ---- fragments (A: 16x32 f16; lane<16: K 0..15, lane>=16: K 16..31) ----
    FragH axr[2], axi[2];
#pragma unroll
    for (int i = 0; i < 2; ++i) {
      const int arow = wm * 32 + i * 16 + l15;
      const int kb   = lhi * 16;
      const uint4* p0 = (const uint4*)&lA[0][arow][kb];
      axr[i].q[0] = p0[0]; axr[i].q[1] = p0[1];          // +16B chunk
      const uint4* p1 = (const uint4*)&lA[1][arow][kb];
      axi[i].q[0] = p1[0]; axi[i].q[1] = p1[1];
    }
    // B fragments (32x16: lane<16 col=lane K 0..15; lane>=16 col=lane-16 K 16..31)
    FragH bwr[4], bwi[4], bwn[4];
#pragma unroll
    for (int j = 0; j < 4; ++j) {
      const int brow = wn * 64 + j * 16 + l15;
      const int kb   = lhi * 16;
      const uint4* p0 = (const uint4*)&lB[0][brow][kb];
      bwr[j].q[0] = p0[0]; bwr[j].q[1] = p0[1];
      const uint4* p1 = (const uint4*)&lB[1][brow][kb];
      bwi[j].q[0] = p1[0]; bwi[j].q[1] = p1[1];
      const uint4* p2 = (const uint4*)&lB[2][brow][kb];
      bwn[j].q[0] = p2[0]; bwn[j].q[1] = p2[1];
    }

    // ---- 32 WMMAs per K-step, no VALU in between ----
#pragma unroll
    for (int i = 0; i < 2; ++i)
#pragma unroll
      for (int j = 0; j < 4; ++j) {
        accR[i][j] = __builtin_amdgcn_wmma_f32_16x16x32_f16(false, axr[i].v, false, bwr[j].v,
                                                            (short)0, accR[i][j], false, false);
        accR[i][j] = __builtin_amdgcn_wmma_f32_16x16x32_f16(false, axi[i].v, false, bwn[j].v,
                                                            (short)0, accR[i][j], false, false);
        accI[i][j] = __builtin_amdgcn_wmma_f32_16x16x32_f16(false, axr[i].v, false, bwi[j].v,
                                                            (short)0, accI[i][j], false, false);
        accI[i][j] = __builtin_amdgcn_wmma_f32_16x16x32_f16(false, axi[i].v, false, bwr[j].v,
                                                            (short)0, accI[i][j], false, false);
      }
  }

  // ---- writeback: C/D layout — VGPR e: lanes0-15 M=e, lanes16-31 M=e+8; N=lane&15 ----
  float* __restrict__ yre = out;
  float* __restrict__ yim = out + (size_t)M_TOK * N_OUT;
#pragma unroll
  for (int i = 0; i < 2; ++i)
#pragma unroll
    for (int j = 0; j < 4; ++j) {
      const int row0 = bm + wm * 32 + i * 16 + lhi * 8;
      const int col  = bn + wn * 64 + j * 16 + l15;
#pragma unroll
      for (int e = 0; e < 8; ++e) {
        yre[(size_t)(row0 + e) * N_OUT + col] = accR[i][j][e];
        yim[(size_t)(row0 + e) * N_OUT + col] = accI[i][j][e];
      }
    }
}

// ---------------------------------------------------------------------------
extern "C" void kernel_launch(void* const* d_in, const int* in_sizes, int n_in,
                              void* d_out, int out_size, void* d_ws, size_t ws_size,
                              hipStream_t stream) {
  const float* xre = (const float*)d_in[0];
  const float* xim = (const float*)d_in[1];
  const float* wre = (const float*)d_in[2];
  const float* wim = (const float*)d_in[3];

  float* ws_f     = (float*)d_ws;
  float* partials = ws_f;          // 1024 floats
  float* params   = ws_f + 1024;   // 4 floats

  abssum_kernel<<<512, 256, 0, stream>>>(wre, wim, partials);
  finalize_kernel<<<1, 256, 0, stream>>>(partials, params);

  dim3 grid(N_OUT / BN, M_TOK / BM);   // (32, 64)
  cgemm_q_kernel<<<grid, 256, 0, stream>>>(xre, xim, wre, wim, params, (float*)d_out);
}
